// SegmentCausalCrossAttention_47742856463121
// MI455X (gfx1250) — matrix operand order repaired and checked by reading, not compiled, not run on hardware
//
#include <hip/hip_runtime.h>
#include <hip/hip_bf16.h>
#include <math.h>

// Problem constants (from reference)
#define BB     4
#define LQ     4096
#define LKV    512
#define DMODEL 512
#define NHEAD  8
#define DHEAD  64
#define KW     9          // LOOKBACK + 1
#define SCALE  0.125f     // 64^-0.5

typedef _Float16 half4  __attribute__((ext_vector_type(4)));
typedef _Float16 half8  __attribute__((ext_vector_type(8)));
typedef _Float16 v16h   __attribute__((ext_vector_type(16)));
typedef float    v8f    __attribute__((ext_vector_type(8)));

// ---------------------------------------------------------------------------
// f32 -> f16 conversion, 4 elems/thread
// ---------------------------------------------------------------------------
__global__ __launch_bounds__(256)
void cvt_f32_to_f16_kernel(const float* __restrict__ x, _Float16* __restrict__ y, int n4) {
    int i = blockIdx.x * blockDim.x + threadIdx.x;
    if (i < n4) {
        float4 v = ((const float4*)x)[i];
        half4 o;
        o[0] = (_Float16)v.x; o[1] = (_Float16)v.y;
        o[2] = (_Float16)v.z; o[3] = (_Float16)v.w;
        ((half4*)y)[i] = o;
    }
}

// ---------------------------------------------------------------------------
// W (512x512 f32 row-major) -> W^T (512x512 f16 row-major, i.e. WT[n][k]=W[k][n])
// ---------------------------------------------------------------------------
__global__ __launch_bounds__(256)
void transpose_w_f16_kernel(const float* __restrict__ W, _Float16* __restrict__ WT) {
    int idx = blockIdx.x * blockDim.x + threadIdx.x;   // 512*512 threads
    int k = idx >> 9;
    int n = idx & 511;
    WT[(size_t)n * DMODEL + k] = (_Float16)W[idx];
}

// ---------------------------------------------------------------------------
// GEMM: C(MxN) = A(MxK,f16) * B(KxN) + bias, N=K=512, via WMMA f32_16x16x32_f16.
// B supplied transposed (BT[n][k], f16). Each wave computes a 32x64 strip
// (2 M-tiles x 4 N-tiles): per k-step 12 b128 loads feed 8 WMMAs.
// HAS_MASK variant applies a per-row zero mask (q_pad_mask) branch-free.
// ---------------------------------------------------------------------------
template <bool HAS_MASK>
__global__ __launch_bounds__(128)
void gemm_f16_wmma_kernel(const _Float16* __restrict__ A,
                          const _Float16* __restrict__ BT,
                          const float* __restrict__ bias,
                          const unsigned char* __restrict__ row_mask,  // used iff HAS_MASK
                          float* __restrict__ C,
                          int M) {
    const int K = DMODEL, N = DMODEL;
    int wave = (blockIdx.x * blockDim.x + threadIdx.x) >> 5;
    int lane = threadIdx.x & 31;
    int ngroups = N / 64;                 // 8 strips of 64 cols
    int m0 = (wave / ngroups) * 32;       // 32 rows per wave
    int n0 = (wave % ngroups) * 64;
    if (m0 >= M) return;                  // wave-uniform (EXEC stays full)

    int l16  = lane & 15;
    int half = lane >> 4;

    v8f acc[2][4] = {};

#pragma unroll 2
    for (int k0 = 0; k0 < K; k0 += 32) {
        // A fragments (two 16x32 f16 tiles): lane holds row m0 + u*16 + l16,
        //   elems 0..7  -> K = k0 + 16*half + 0..7
        //   elems 8..15 -> K = k0 + 16*half + 8..15
        v16h afrag[2];
#pragma unroll
        for (int u = 0; u < 2; ++u) {
            const _Float16* ap = A + (size_t)(m0 + u * 16 + l16) * K + k0 + 16 * half;
            half8 alo = *(const half8*)ap;
            half8 ahi = *(const half8*)(ap + 8);
#pragma unroll
            for (int i = 0; i < 8; ++i) { afrag[u][i] = alo[i]; afrag[u][i + 8] = ahi[i]; }
        }

        // B fragments (32x16 f16): lane holds col n0+t*16+l16, K = k0 + 16*half + e
#pragma unroll
        for (int t = 0; t < 4; ++t) {
            const _Float16* bp = BT + (size_t)(n0 + t * 16 + l16) * K + k0 + 16 * half;
            v16h bfrag = *(const v16h*)bp;
#pragma unroll
            for (int u = 0; u < 2; ++u) {
                acc[u][t] = __builtin_amdgcn_wmma_f32_16x16x32_f16(
                    false, afrag[u], false, bfrag, (short)0, acc[u][t], false, false);
            }
        }
    }

    // Per-row keep factor (branch-free): keep = 1 - mask  (mask in {0,1})
    float keep[2][8];
    if (HAS_MASK) {
#pragma unroll
        for (int u = 0; u < 2; ++u)
#pragma unroll
            for (int r = 0; r < 8; ++r)
                keep[u][r] = 1.0f - (float)row_mask[m0 + u * 16 + r + 8 * half];
    }

    // Epilogue: C/D layout — lane l16 = col, VGPR r -> row m0 + u*16 + r + 8*half
#pragma unroll
    for (int u = 0; u < 2; ++u) {
#pragma unroll
        for (int t = 0; t < 4; ++t) {
            int col = n0 + t * 16 + l16;
            float bv = bias[col];
#pragma unroll
            for (int r = 0; r < 8; ++r) {
                int mrow = m0 + u * 16 + r + 8 * half;
                float val = acc[u][t][r] + bv;
                if (HAS_MASK) val *= keep[u][r];
                C[(size_t)mrow * N + col] = val;
            }
        }
    }
}

// ---------------------------------------------------------------------------
// Windowed attention: one wave per (b, l, h). Lanes split DH=64 (elems lane,
// lane+32). 9 gathered dot products, softmax over 9, weighted V sum.
// ---------------------------------------------------------------------------
__global__ __launch_bounds__(256)
void attn_window_kernel(const float* __restrict__ qh,     // (B*LQ, 512)
                        const float* __restrict__ kh,     // (B*LKV, 512)
                        const float* __restrict__ vh,     // (B*LKV, 512)
                        const int* __restrict__ seg_id,   // (B*LQ)
                        const unsigned char* __restrict__ kv_mask, // (B*LKV)
                        float* __restrict__ attn_out) {   // (B*LQ, 512)
    int lane = threadIdx.x & 31;
    int task = (blockIdx.x * blockDim.x + threadIdx.x) >> 5;  // (b*LQ + l)*H + h
    int h  = task & (NHEAD - 1);
    int bl = task >> 3;         // b*LQ + l
    int b  = bl >> 12;          // / LQ

    int seg = seg_id[bl];
    const float* qrow = qh + (size_t)bl * DMODEL + h * DHEAD;
    float q0 = qrow[lane];
    float q1 = qrow[lane + 32];

    float sc[KW];
    int   idxs[KW];
#pragma unroll
    for (int w = 0; w < KW; ++w) {
        int idx = seg - w;
        idx = idx < 0 ? 0 : (idx > LKV - 1 ? LKV - 1 : idx);
        idxs[w] = idx;
        const float* krow = kh + (size_t)(b * LKV + idx) * DMODEL + h * DHEAD;
        float p = q0 * krow[lane] + q1 * krow[lane + 32];
#pragma unroll
        for (int off = 16; off > 0; off >>= 1) p += __shfl_xor(p, off, 32);
        sc[w] = kv_mask[b * LKV + idx] ? -3.402823466e38f : p * SCALE;
    }

    float m = sc[0];
#pragma unroll
    for (int w = 1; w < KW; ++w) m = fmaxf(m, sc[w]);
    float pr[KW], s = 0.0f;
#pragma unroll
    for (int w = 0; w < KW; ++w) { pr[w] = __expf(sc[w] - m); s += pr[w]; }
    float inv = 1.0f / s;

    float o0 = 0.0f, o1 = 0.0f;
#pragma unroll
    for (int w = 0; w < KW; ++w) {
        const float* vrow = vh + (size_t)(b * LKV + idxs[w]) * DMODEL + h * DHEAD;
        float p = pr[w] * inv;
        o0 += p * vrow[lane];
        o1 += p * vrow[lane + 32];
    }
    float* orow = attn_out + (size_t)bl * DMODEL + h * DHEAD;
    orow[lane]      = o0;
    orow[lane + 32] = o1;
}

// ---------------------------------------------------------------------------
// Host driver
// ---------------------------------------------------------------------------
extern "C" void kernel_launch(void* const* d_in, const int* in_sizes, int n_in,
                              void* d_out, int out_size, void* d_ws, size_t ws_size,
                              hipStream_t stream) {
    const float* q      = (const float*)d_in[0];
    const float* kv_src = (const float*)d_in[1];
    const int*   seg_id = (const int*)d_in[2];
    const unsigned char* kv_mask    = (const unsigned char*)d_in[3];
    const unsigned char* q_pad_mask = (const unsigned char*)d_in[4];
    const float* Wq = (const float*)d_in[5];
    const float* bq = (const float*)d_in[6];
    const float* Wk = (const float*)d_in[7];
    const float* bk = (const float*)d_in[8];
    const float* Wv = (const float*)d_in[9];
    const float* bv = (const float*)d_in[10];
    const float* Wo = (const float*)d_in[11];
    const float* bo = (const float*)d_in[12];
    float* out = (float*)d_out;

    const int nq  = BB * LQ * DMODEL;     // 8,388,608
    const int nkv = BB * LKV * DMODEL;    // 1,048,576
    const int nw  = DMODEL * DMODEL;      // 262,144

    // Workspace carve-up (~92 MB)
    char* ws = (char*)d_ws;
    size_t off = 0;
    _Float16* qf16  = (_Float16*)(ws + off); off += (size_t)nq * 2;     // 16MB (reused for attn f16)
    _Float16* kvf16 = (_Float16*)(ws + off); off += (size_t)nkv * 2;    // 2MB
    _Float16* WqT   = (_Float16*)(ws + off); off += (size_t)nw * 2;
    _Float16* WkT   = (_Float16*)(ws + off); off += (size_t)nw * 2;
    _Float16* WvT   = (_Float16*)(ws + off); off += (size_t)nw * 2;
    _Float16* WoT   = (_Float16*)(ws + off); off += (size_t)nw * 2;
    float* qh_buf   = (float*)(ws + off);    off += (size_t)nq * 4;     // 32MB
    float* kh_buf   = (float*)(ws + off);    off += (size_t)nkv * 4;    // 4MB
    float* vh_buf   = (float*)(ws + off);    off += (size_t)nkv * 4;    // 4MB
    float* attn_buf = (float*)(ws + off);    off += (size_t)nq * 4;     // 32MB

    // 1) Convert activations + weights to f16 (weights transposed)
    cvt_f32_to_f16_kernel<<<(nq / 4 + 255) / 256, 256, 0, stream>>>(q, qf16, nq / 4);
    cvt_f32_to_f16_kernel<<<(nkv / 4 + 255) / 256, 256, 0, stream>>>(kv_src, kvf16, nkv / 4);
    transpose_w_f16_kernel<<<nw / 256, 256, 0, stream>>>(Wq, WqT);
    transpose_w_f16_kernel<<<nw / 256, 256, 0, stream>>>(Wk, WkT);
    transpose_w_f16_kernel<<<nw / 256, 256, 0, stream>>>(Wv, WvT);
    transpose_w_f16_kernel<<<nw / 256, 256, 0, stream>>>(Wo, WoT);

    // 2) Projections (WMMA). waves = (M/32)*8 ; blocks = waves/4 (128 thr = 4 waves)
    const int Mq  = BB * LQ;    // 16384 -> 1024 blocks
    const int Mkv = BB * LKV;   // 2048  -> 128 blocks
    gemm_f16_wmma_kernel<false><<<(Mq / 32) * 8 / 4, 128, 0, stream>>>(qf16, WqT, bq, nullptr, qh_buf, Mq);
    gemm_f16_wmma_kernel<false><<<(Mkv / 32) * 8 / 4, 128, 0, stream>>>(kvf16, WkT, bk, nullptr, kh_buf, Mkv);
    gemm_f16_wmma_kernel<false><<<(Mkv / 32) * 8 / 4, 128, 0, stream>>>(kvf16, WvT, bv, nullptr, vh_buf, Mkv);

    // 3) Windowed attention: one wave per (b,l,h) -> B*LQ*H waves, 8 waves/block
    const int attn_waves = BB * LQ * NHEAD;      // 131072
    attn_window_kernel<<<attn_waves / 8, 256, 0, stream>>>(qh_buf, kh_buf, vh_buf,
                                                           seg_id, kv_mask, attn_buf);

    // 4) Output projection with bias + q_pad_mask zeroing (reuse qf16 for f16 attn)
    cvt_f32_to_f16_kernel<<<(nq / 4 + 255) / 256, 256, 0, stream>>>(attn_buf, qf16, nq / 4);
    gemm_f16_wmma_kernel<true><<<(Mq / 32) * 8 / 4, 128, 0, stream>>>(qf16, WoT, bo, q_pad_mask, out, Mq);

    (void)in_sizes; (void)n_in; (void)out_size; (void)ws_size;
}
